// PointNet2PatchletsSA_6957847020167
// MI455X (gfx1250) — matrix operand order repaired and checked by reading, not compiled, not run on hardware
//
#include <hip/hip_runtime.h>

// ---------------------------------------------------------------------------
// PointNet2-Patchlets SA for MI455X (gfx1250, wave32, WMMA).
// Conv layers as im2col GEMMs on v_wmma_f32_16x16x32_f16; double-buffered,
// fully branchless LDS staging (padded-segment shifted stores).
// ---------------------------------------------------------------------------

typedef __attribute__((ext_vector_type(16))) _Float16 v16h;
typedef __attribute__((ext_vector_type(8)))  _Float16 v8h;
typedef __attribute__((ext_vector_type(8)))  float    v8f;
typedef __attribute__((ext_vector_type(4)))  unsigned v4u;

#define B_   4
#define T_   64
#define N_   1024
#define K_   16
#define SPAT 3072            // 3 * 64 * 16 spatial positions per (b, channel)
#define AW   72              // As col-row stride in halves (144B)
#define BSEG 32              // halves per (col, dt) segment; real window [8..24)
#define BW   136             // Bs col-row stride in halves (272B: 16B-aligned, 4-bank step)

// ============================== kNN ========================================
__global__ __launch_bounds__(256) void knn_kernel(const float* __restrict__ xyz,
                                                  int* __restrict__ idxs) {
  __shared__ float dbx[N_], dby[N_], dbz[N_];
  const int f = blockIdx.x;
  const int b = f >> 6, t = f & 63;
  const int tid = threadIdx.x;
  const float* xb = xyz + (size_t)f * N_ * 3;
  const int tprev = (t == 0) ? 0 : (t - 1);
  const float* xq = xyz + ((size_t)(b * T_ + tprev)) * N_ * 3;
  for (int i = tid; i < N_; i += 256) {
    dbx[i] = xb[i * 3 + 0]; dby[i] = xb[i * 3 + 1]; dbz[i] = xb[i * 3 + 2];
  }
  __syncthreads();
  const int q = blockIdx.y * 256 + tid;
  const float qx = xq[q * 3 + 0], qy = xq[q * 3 + 1], qz = xq[q * 3 + 2];
  float bd[K_]; int bi[K_];
#pragma unroll
  for (int j = 0; j < K_; ++j) { bd[j] = 3.4e38f; bi[j] = 0; }
  for (int j = 0; j < N_; ++j) {
    const float dx = qx - dbx[j], dy = qy - dby[j], dz = qz - dbz[j];
    const float d2 = dx * dx + dy * dy + dz * dz;
    if (d2 < bd[K_ - 1]) {
      bd[K_ - 1] = d2; bi[K_ - 1] = j;
#pragma unroll
      for (int s = K_ - 1; s > 0; --s) {
        if (bd[s] < bd[s - 1]) {
          const float tf = bd[s]; bd[s] = bd[s - 1]; bd[s - 1] = tf;
          const int   ti = bi[s]; bi[s] = bi[s - 1]; bi[s - 1] = ti;
        }
      }
    }
  }
  int* out = idxs + ((size_t)f * N_ + q) * K_;
#pragma unroll
  for (int j = 0; j < K_; ++j) out[j] = bi[j];
}

// ===================== patchlet seed propagation (scan) ====================
__global__ void propagate_kernel(const int* __restrict__ idxs,
                                 int* __restrict__ patch) {
  const int g = blockIdx.x * blockDim.x + threadIdx.x;   // 4096 chains
  const int b = g >> 10, n = g & 1023;
  int carry = n;
  for (int t = 0; t < T_; ++t) {
    const int* row  = idxs  + (((size_t)(b * T_ + t)) * N_ + carry) * K_;
    int*       orow = patch + (((size_t)(b * T_ + t)) * N_ + n) * K_;
    const int first = row[0];
#pragma unroll
    for (int k = 0; k < K_; ++k) orow[k] = row[k];
    carry = first;
  }
}

// ====================== gather + pack (f32 -> f16 NCDHW) ===================
__global__ void gather_pack_kernel(const float* __restrict__ xyz,
                                   const int* __restrict__ patch,
                                   _Float16* __restrict__ act0,
                                   float* __restrict__ out_xyz) {
  const size_t g = (size_t)blockIdx.x * 256 + threadIdx.x;   // B*T*N*K
  if (g >= (size_t)B_ * T_ * N_ * K_) return;
  const int k = (int)(g & 15);
  size_t r = g >> 4;
  const int n = (int)(r & 1023); r >>= 10;
  const int t = (int)(r & 63);
  const int b = (int)(r >> 6);
  const int j = patch[g];
  const float* p = xyz + (((size_t)(b * T_ + t)) * N_ + j) * 3;
  const float x = p[0], y = p[1], z = p[2];
  const size_t base = (((size_t)b * N_ + n) * 3) * 1024 + t * 16 + k;
  act0[base]        = (_Float16)x;
  act0[base + 1024] = (_Float16)y;
  act0[base + 2048] = (_Float16)z;
  if (k == 0) {
    const size_t ob = (((size_t)(b * T_ + t)) * 3) * N_ + n;
    out_xyz[ob]          = x;
    out_xyz[ob + N_]     = y;
    out_xyz[ob + 2 * N_] = z;
  }
}

// ====================== weight f32 -> f16 repack ===========================
__global__ void wpack_kernel(const float* __restrict__ w,
                             _Float16* __restrict__ wp, int count) {
  const int idx = blockIdx.x * 256 + threadIdx.x;
  if (idx < count) wp[idx] = (_Float16)w[idx];
}

// ============================ conv as WMMA GEMM ============================
// Block = 256 threads (8 wave32) -> 64(M) x 64(N) tile; double-buffered LDS.
// B staging is branchless: each (col, dt) owns a 32-half padded segment with
// the real 16-half window at [8..24); the k-axis shifted write 8+(kp+7-kk)
// always lands in [0,31], so all 16 ds_store_b16 are unconditional.
// The t-halo is handled by a loop-invariant AND mask on a clamped load.
__global__ __launch_bounds__(256) void conv_gemm(const _Float16* __restrict__ act,
                                                 const _Float16* __restrict__ wp,
                                                 float* __restrict__ out,
                                                 int Cin, int Cout) {
  __shared__ __align__(16) _Float16 As[2][64 * AW];
  __shared__ __align__(16) _Float16 Bs[2][64 * BW];
  const int tid = threadIdx.x;
  const int co_base = blockIdx.y * 64;
  const int ntile = blockIdx.x;              // 0..191
  const int bd = ntile >> 4;                 // b*3 + d
  const int b = bd / 3, d = bd % 3;
  const int t0 = (ntile & 15) * 4;
  const int lane = tid & 31, lane15 = lane & 15, hi = lane >> 4;
  const int wv = tid >> 5;
  const int mt = wv & 3;
  const int nt1 = wv >> 2, nt2 = (wv >> 2) + 2;
  v8f c0 = {}, c1 = {};

  // ---- staging roles ----
  const int row_a = tid >> 2;                // 0..63 weight row
  const int qa    = (tid & 3) * 16;          // tap quarter
  const int dt_b  = tid >> 6;                // 0..3 kernel t-tap
  const int col_b = tid & 63;                // 0..63 output column
  const int tt_b  = col_b >> 4, kk_b = col_b & 15;
  const int toff  = t0 + tt_b + dt_b - 1;    // t-pad (1,2) for kernel 4
  const int tcl   = toff < 0 ? 0 : (toff > 63 ? 63 : toff);   // always-valid addr
  const unsigned mk = ((unsigned)toff < 64u) ? 0xFFFFFFFFu : 0u;
  const v4u vmask = {mk, mk, mk, mk};        // loop-invariant halo mask
  const _Float16* wbase = wp + ((size_t)(co_base + row_a) * Cin) * 64 + qa;
  const _Float16* abase = act + ((((size_t)b * Cin) * 3 + d) * 1024) + tcl * 16;

  v8h pa0, pa1, pb0, pb1;
  auto LOAD = [&](int ci) {
    const _Float16* wpc = wbase + (size_t)ci * 64;
    pa0 = *(const v8h*)(wpc);
    pa1 = *(const v8h*)(wpc + 8);
    const _Float16* ap = abase + (size_t)ci * SPAT;
    const v8h r0 = *(const v8h*)(ap);
    const v8h r1 = *(const v8h*)(ap + 8);
    pb0 = (v8h)(((v4u)r0) & vmask);          // zero halo rows, no branch
    pb1 = (v8h)(((v4u)r1) & vmask);
  };
  auto STORE = [&](int buf) {
    _Float16* adst = &As[buf][row_a * AW + qa];
    *(v8h*)(adst)     = pa0;
    *(v8h*)(adst + 8) = pa1;
    _Float16* seg = &Bs[buf][col_b * BW + dt_b * BSEG];
    const v8h z = {};
    *(v8h*)(seg + 8)  = z;                   // zero real window only
    *(v8h*)(seg + 16) = z;
    _Float16 row[16];
#pragma unroll
    for (int kp = 0; kp < 8; ++kp) { row[kp] = pb0[kp]; row[kp + 8] = pb1[kp]; }
    const int sh = 15 - kk_b;                // 8 + (kp + 7 - kk): in [0,31] always
#pragma unroll
    for (int kp = 0; kp < 16; ++kp) seg[sh + kp] = row[kp];   // branchless
  };

  LOAD(0);
  STORE(0);
  __syncthreads();

  for (int ci = 0; ci < Cin; ++ci) {
    const int cur = ci & 1;
    const bool more = (ci + 1) < Cin;
    if (more) LOAD(ci + 1);                  // global loads overlap WMMAs below

    const _Float16* arow = &As[cur][(mt * 16 + lane15) * AW];
    const _Float16* brow1 = &Bs[cur][(nt1 * 16 + lane15) * BW];
    const _Float16* brow2 = &Bs[cur][(nt2 * 16 + lane15) * BW];
#pragma unroll
    for (int ks = 0; ks < 2; ++ks) {
      // A 16x32 f16: lanes 0-15 -> K {0..15}, lanes 16-31 -> {16..31}
      const int kbA = ks * 32 + (hi ? 16 : 0);
      const v8h alo = *(const v8h*)(arow + kbA);
      const v8h ahi = *(const v8h*)(arow + kbA + 8);
      const v16h a = __builtin_shufflevector(alo, ahi,
          0, 1, 2, 3, 4, 5, 6, 7, 8, 9, 10, 11, 12, 13, 14, 15);
      // B fragment = one dt segment: taps [kbB, kbB+16) == dt = ks*2 + hi
      const int soff = (ks * 2 + hi) * BSEG + 8;
      const v8h b0l = *(const v8h*)(brow1 + soff);
      const v8h b0h = *(const v8h*)(brow1 + soff + 8);
      const v16h bf0 = __builtin_shufflevector(b0l, b0h,
          0, 1, 2, 3, 4, 5, 6, 7, 8, 9, 10, 11, 12, 13, 14, 15);
      const v8h b1l = *(const v8h*)(brow2 + soff);
      const v8h b1h = *(const v8h*)(brow2 + soff + 8);
      const v16h bf1 = __builtin_shufflevector(b1l, b1h,
          0, 1, 2, 3, 4, 5, 6, 7, 8, 9, 10, 11, 12, 13, 14, 15);
      c0 = __builtin_amdgcn_wmma_f32_16x16x32_f16(false, a, false, bf0,
                                                  (short)0, c0, false, false);
      c1 = __builtin_amdgcn_wmma_f32_16x16x32_f16(false, a, false, bf1,
                                                  (short)0, c1, false, false);
    }
    if (more) STORE(cur ^ 1);                // fill alternate buffer
    __syncthreads();
  }

  // ---- epilogue: C/D layout VGPR r -> M = r + 8*hi, lane15 -> N ----
  const size_t outB = (((size_t)b * Cout + co_base) * 3 + d) * 1024;
#pragma unroll
  for (int r = 0; r < 8; ++r) {
    const int co_l = mt * 16 + r + hi * 8;
    {
      const int col = nt1 * 16 + lane15;
      const int ti = t0 + (col >> 4), kk = col & 15;
      out[outB + (size_t)co_l * SPAT + ti * 16 + kk] = c0[r];
    }
    {
      const int col = nt2 * 16 + lane15;
      const int ti = t0 + (col >> 4), kk = col & 15;
      out[outB + (size_t)co_l * SPAT + ti * 16 + kk] = c1[r];
    }
  }
}

// ====================== BatchNorm (training-mode stats) ====================
__global__ __launch_bounds__(256) void bn_stats(const float* __restrict__ y,
                                                float* __restrict__ mean,
                                                float* __restrict__ rstd, int Cout) {
  __shared__ float s0[256], s1[256];
  const int c = blockIdx.x, tid = threadIdx.x;
  float sum = 0.f, sq = 0.f;
  for (int e = tid; e < B_ * SPAT; e += 256) {
    const int bb = e / SPAT, i = e - bb * SPAT;
    const float v = y[((size_t)bb * Cout + c) * SPAT + i];
    sum += v; sq += v * v;
  }
  s0[tid] = sum; s1[tid] = sq;
  __syncthreads();
  for (int off = 128; off > 0; off >>= 1) {
    if (tid < off) { s0[tid] += s0[tid + off]; s1[tid] += s1[tid + off]; }
    __syncthreads();
  }
  if (tid == 0) {
    const float inv = 1.f / (float)(B_ * SPAT);
    const float m = s0[0] * inv;
    const float var = s1[0] * inv - m * m;     // biased var (jnp.var default)
    mean[c] = m;
    rstd[c] = rsqrtf(var + 1e-5f);
  }
}

// norm + gamma/beta + ReLU, repack to f16 (conv bias cancels under BN).
__global__ void bn_apply_pack(const float* __restrict__ y,
                              const float* __restrict__ mean,
                              const float* __restrict__ rstd,
                              const float* __restrict__ gamma,
                              const float* __restrict__ beta,
                              _Float16* __restrict__ actout, int Cout) {
  const size_t idx = (size_t)blockIdx.x * 256 + threadIdx.x;
  if (idx >= (size_t)B_ * Cout * SPAT) return;
  const int c = (int)((idx / SPAT) % Cout);
  const float v = (y[idx] - mean[c]) * rstd[c] * gamma[c] + beta[c];
  actout[idx] = (_Float16)fmaxf(v, 0.f);
}

// final layer: norm + ReLU + max over k; feats[b][t][c][d] (flat = idx order).
__global__ void bn_max_feats(const float* __restrict__ y,
                             const float* __restrict__ mean,
                             const float* __restrict__ rstd,
                             const float* __restrict__ gamma,
                             const float* __restrict__ beta,
                             float* __restrict__ feats) {
  const int idx = blockIdx.x * 256 + threadIdx.x;   // B*T*256*3
  const int d = idx % 3;
  const int c = (idx / 3) & 255;
  const int t = (idx / 768) & 63;
  const int b = idx / 49152;
  const float m = mean[c], r = rstd[c], g = gamma[c], be = beta[c];
  const float* src = y + (((size_t)b * 256 + c) * 3 + d) * 1024 + t * 16;
  float mx = 0.f;
#pragma unroll
  for (int k = 0; k < 16; ++k)
    mx = fmaxf(mx, fmaxf((src[k] - m) * r * g + be, 0.f));
  feats[idx] = mx;
}

// ================================ driver ===================================
extern "C" void kernel_launch(void* const* d_in, const int* in_sizes, int n_in,
                              void* d_out, int out_size, void* d_ws, size_t ws_size,
                              hipStream_t stream) {
  (void)in_sizes; (void)n_in; (void)out_size; (void)ws_size;
  const float* xyz = (const float*)d_in[0];
  const float* w0  = (const float*)d_in[1];
  const float* g0  = (const float*)d_in[3];
  const float* be0 = (const float*)d_in[4];
  const float* w1  = (const float*)d_in[5];
  const float* g1  = (const float*)d_in[7];
  const float* be1 = (const float*)d_in[8];
  const float* w2  = (const float*)d_in[9];
  const float* g2  = (const float*)d_in[11];
  const float* be2 = (const float*)d_in[12];
  float* out_xyz = (float*)d_out;                 // (b,t,d,n): 786432 floats
  float* feats   = (float*)d_out + 786432;        // (b,t,256,3): 196608 floats

  char* ws = (char*)d_ws;
  int*      idxs  = (int*)(ws);                           // 16 MiB
  int*      patch = (int*)(ws + (16ull << 20));           // 16 MiB
  _Float16* act0  = (_Float16*)(ws + (32ull << 20));      // 24 MiB
  _Float16* wp0   = (_Float16*)(ws + (56ull << 20));      // 8 MiB
  _Float16* wp1   = (_Float16*)(ws + (64ull << 20));      // 1 MiB
  _Float16* wp2   = (_Float16*)(ws + (65ull << 20));      // 4 MiB
  float*    stats = (float*)(ws + (69ull << 20));         // 12 KiB
  float *mean0 = stats,        *rstd0 = stats + 512;
  float *mean1 = stats + 1024, *rstd1 = stats + 1536;
  float *mean2 = stats + 2048, *rstd2 = stats + 2560;
  // reuse: idxs region after propagate; patch region after gather
  float*    conv1 = (float*)(ws);                         // 3 MiB
  _Float16* act1  = (_Float16*)(ws + (3ull  << 20));      // 1.5 MiB
  float*    conv2 = (float*)(ws + (5ull  << 20));         // 6 MiB
  _Float16* act2  = (_Float16*)(ws + (11ull << 20));      // 3 MiB
  float*    conv3 = (float*)(ws + (16ull << 20));         // 12 MiB

  knn_kernel<<<dim3(B_ * T_, 4), 256, 0, stream>>>(xyz, idxs);
  propagate_kernel<<<16, 256, 0, stream>>>(idxs, patch);
  wpack_kernel<<<(64 * 1024 * 64) / 256, 256, 0, stream>>>(w0, wp0, 64 * 1024 * 64);
  wpack_kernel<<<(128 * 64 * 64) / 256, 256, 0, stream>>>(w1, wp1, 128 * 64 * 64);
  wpack_kernel<<<(256 * 128 * 64) / 256, 256, 0, stream>>>(w2, wp2, 256 * 128 * 64);
  gather_pack_kernel<<<(B_ * T_ * N_ * K_) / 256, 256, 0, stream>>>(xyz, patch, act0, out_xyz);

  conv_gemm<<<dim3(192, 1), 256, 0, stream>>>(act0, wp0, conv1, 1024, 64);
  bn_stats<<<64, 256, 0, stream>>>(conv1, mean0, rstd0, 64);
  bn_apply_pack<<<(B_ * 64 * SPAT) / 256, 256, 0, stream>>>(conv1, mean0, rstd0, g0, be0, act1, 64);

  conv_gemm<<<dim3(192, 2), 256, 0, stream>>>(act1, wp1, conv2, 64, 128);
  bn_stats<<<128, 256, 0, stream>>>(conv2, mean1, rstd1, 128);
  bn_apply_pack<<<(B_ * 128 * SPAT) / 256, 256, 0, stream>>>(conv2, mean1, rstd1, g1, be1, act2, 128);

  conv_gemm<<<dim3(192, 4), 256, 0, stream>>>(act2, wp2, conv3, 128, 256);
  bn_stats<<<256, 256, 0, stream>>>(conv3, mean2, rstd2, 256);
  bn_max_feats<<<(B_ * T_ * 256 * 3) / 256, 256, 0, stream>>>(conv3, mean2, rstd2, g2, be2, feats);
}
